// ModulateMLP_56453050139463
// MI455X (gfx1250) — matrix operand order repaired and checked by reading, not compiled, not run on hardware
//
#include <hip/hip_runtime.h>

typedef __attribute__((ext_vector_type(16))) _Float16 v16h;
typedef __attribute__((ext_vector_type(8)))  float    v8f;
typedef __attribute__((ext_vector_type(4)))  unsigned int u32x4;
typedef __attribute__((ext_vector_type(8)))  int i32x8;
typedef __attribute__((ext_vector_type(4)))  int i32x4;

#define IN_CH  512
#define OUT_CH 512
#define WDIM   512
#define SEQ    16384
#define LDSP   36   // padded LDS row stride in f32 (32 data + 4 pad)

#if defined(__gfx1250__) && __has_builtin(__builtin_amdgcn_tensor_load_to_lds) && __has_builtin(__builtin_amdgcn_s_wait_tensorcnt)
#define USE_TDM 1
#else
#define USE_TDM 0
#endif

// ---------------- style = w @ style_w^T * wscale + style_b ----------------
__global__ __launch_bounds__(512) void style_kernel(const float* __restrict__ w,
                                                    const float* __restrict__ sw,
                                                    const float* __restrict__ sb,
                                                    float* __restrict__ style) {
  const int n = blockIdx.x;
  const int c = threadIdx.x;
  const float4* wr  = (const float4*)(w + (size_t)n * WDIM);
  const float4* swr = (const float4*)(sw + (size_t)c * WDIM);
  float s = 0.f;
#pragma unroll 4
  for (int i = 0; i < WDIM / 4; ++i) {
    float4 a = wr[i], b = swr[i];
    s += a.x * b.x + a.y * b.y + a.z * b.z + a.w * b.w;
  }
  style[(size_t)n * IN_CH + c] = s * 0.04419417382415922f + sb[c];
}

// ----- W'[n,o,c] = weight[o,c]*style[n,c]*rsqrt(sum_c (weight*style)^2 + eps), f16 -----
__global__ __launch_bounds__(256) void modw_kernel(const float* __restrict__ weight,
                                                   const float* __restrict__ style,
                                                   _Float16* __restrict__ wmod) {
  const int o = blockIdx.x;
  const int n = blockIdx.y;
  const int t = threadIdx.x;
  const float* wr = weight + (size_t)o * IN_CH;
  const float* st = style + (size_t)n * IN_CH;
  float p0 = wr[t] * st[t];
  float p1 = wr[t + 256] * st[t + 256];
  __shared__ float red[256];
  red[t] = p0 * p0 + p1 * p1;
  __syncthreads();
  for (int s = 128; s > 0; s >>= 1) {
    if (t < s) red[t] += red[t + s];
    __syncthreads();
  }
  float d = rsqrtf(red[0] + 1e-8f);
  _Float16* outp = wmod + ((size_t)n * OUT_CH + o) * IN_CH;
  outp[t]       = (_Float16)(p0 * d);
  outp[t + 256] = (_Float16)(p1 * d);
}

// ---------------- main GEMM: y = x @ W'^T + bias -> lrelu -> clamp ----------------
__global__ __launch_bounds__(256) void gemm_kernel(const float* __restrict__ x,
                                                   const _Float16* __restrict__ wmod,
                                                   const float* __restrict__ bias,
                                                   float* __restrict__ out) {
  __shared__ float As[2][128 * LDSP];

  const int n      = blockIdx.z;
  const int m_base = blockIdx.y * 128;
  const int n_base = blockIdx.x * 128;
  const int tid    = threadIdx.x;
  const int lane   = tid & 31;
  const int wave   = tid >> 5;
  const int waveM  = wave & 1;   // 2 waves over M
  const int waveN  = wave >> 1;  // 4 waves over N
  const int lhalf  = lane >> 4;
  const int l16    = lane & 15;

  const _Float16* wn = wmod + (size_t)n * OUT_CH * IN_CH;
  const float*    xn = x + ((size_t)n * SEQ + m_base) * IN_CH;

  v8f acc[4][2];
#pragma unroll
  for (int i = 0; i < 4; ++i)
#pragma unroll
    for (int j = 0; j < 2; ++j)
#pragma unroll
      for (int r = 0; r < 8; ++r) acc[i][j][r] = 0.f;

  float bb[2];
#pragma unroll
  for (int j = 0; j < 2; ++j) bb[j] = bias[n_base + waveN * 32 + j * 16 + l16];

#if USE_TDM
  // Each wave DMAs its 16-row slice of the 128x32 f32 A-tile (double buffered).
  const unsigned lds_base = (unsigned)(uintptr_t)&As[0][0];
  const unsigned long long x_addr =
      (unsigned long long)(uintptr_t)xn + (unsigned long long)wave * (16ull * IN_CH * 4ull);

  i32x8 g1;
  g1[0] = (int)((2u << 16) | (1u << 20) | (4u << 22) | (3u << 25)); // 4B elems; pad 4DW every 32DW
  g1[1] = (int)(512u << 16);   // tensor_dim0 = 512 (bits 79:48 low half)
  g1[2] = (int)(16384u << 16); // tensor_dim1 = 16384 (bits 111:80 low half)
  g1[3] = (int)(32u << 16);    // tile_dim0 = 32
  g1[4] = 16;                  // tile_dim1 = 16 rows per wave
  g1[5] = 512;                 // tensor_dim0_stride = 512 elems
  g1[6] = 0;
  g1[7] = 0x80;                // tensor_dim1_stride = 512*16384 (bits 255:224)
  const i32x4 gz4 = {0, 0, 0, 0};
  const i32x8 gz8 = {0, 0, 0, 0, 0, 0, 0, 0};

  auto tdm_issue = [&](int buf, int ks) {
    unsigned long long ga = x_addr + (unsigned long long)ks * 128ull; // 32 f32 per K-step
    u32x4 g0;
    g0[0] = 1u; // count = 1 valid descriptor
    g0[1] = lds_base + (unsigned)buf * (unsigned)(128 * LDSP * 4)
                     + (unsigned)wave * (unsigned)(16 * LDSP * 4);
    g0[2] = (unsigned)ga;
    g0[3] = (unsigned)(ga >> 32) | 0x80000000u; // type=2 ("image")
    __builtin_amdgcn_tensor_load_to_lds(g0, g1, gz4, gz4, gz8, 0);
  };

  tdm_issue(0, 0);
#endif

  for (int ks = 0; ks < 16; ++ks) {
#if USE_TDM
    const int buf = ks & 1;
    __builtin_amdgcn_s_wait_tensorcnt(0); // our slice landed in LDS
    __syncthreads();                      // everyone's slice landed; prev buffer free
    if (ks + 1 < 16) tdm_issue(buf ^ 1, ks + 1);
#else
    const int buf = 0;
    __syncthreads();
#pragma unroll
    for (int q = 0; q < 4; ++q) {
      int idx = tid + q * 256; // 1024 float4 slots = 128 rows x 8
      int row = idx >> 3;
      int c4  = idx & 7;
      float4 v = *(const float4*)(xn + (size_t)row * IN_CH + ks * 32 + c4 * 4);
      *(float4*)&As[0][row * LDSP + c4 * 4] = v;
    }
    __syncthreads();
#endif

    // A fragments: 16x32 f16 layout; lanes 0-15 hold K 0..15, lanes 16-31 hold K 16..31.
    v16h afr[4];
#pragma unroll
    for (int i = 0; i < 4; ++i) {
      const float* ap = &As[buf][(waveM * 64 + i * 16 + l16) * LDSP + lhalf * 16];
#pragma unroll
      for (int t = 0; t < 8; ++t) {
        afr[i][t]     = (_Float16)ap[t];
        afr[i][t + 8] = (_Float16)ap[8 + t];
      }
    }
    // B fragments: 32x16 layout, 16 contiguous K per lane from W'[o,:], straight from L2.
    v16h bfr[2];
#pragma unroll
    for (int j = 0; j < 2; ++j) {
      int o = n_base + waveN * 32 + j * 16 + l16;
      bfr[j] = *(const v16h*)(wn + (size_t)o * IN_CH + ks * 32 + lhalf * 16);
    }
#pragma unroll
    for (int i = 0; i < 4; ++i)
#pragma unroll
      for (int j = 0; j < 2; ++j)
        acc[i][j] = __builtin_amdgcn_wmma_f32_16x16x32_f16(
            false, afr[i], false, bfr[j], (short)0, acc[i][j], false, false);
  }

  // epilogue: bias + lrelu(gain=sqrt2) + clamp(+-256); C layout: VGPR r -> M=r(+8 hi half), N=lane&15
#pragma unroll
  for (int i = 0; i < 4; ++i) {
#pragma unroll
    for (int j = 0; j < 2; ++j) {
      const int N = n_base + waveN * 32 + j * 16 + l16;
#pragma unroll
      for (int r = 0; r < 8; ++r) {
        const int M = m_base + waveM * 64 + i * 16 + lhalf * 8 + r;
        float v = acc[i][j][r] + bb[j];
        v = (v >= 0.f ? v : 0.2f * v) * 1.4142135623730951f;
        v = fminf(fmaxf(v, -256.f), 256.f);
        out[((size_t)n * SEQ + M) * OUT_CH + N] = v;
      }
    }
  }
}

extern "C" void kernel_launch(void* const* d_in, const int* in_sizes, int n_in,
                              void* d_out, int out_size, void* d_ws, size_t ws_size,
                              hipStream_t stream) {
  (void)in_sizes; (void)n_in; (void)out_size; (void)ws_size;
  const float* x  = (const float*)d_in[0];
  const float* w  = (const float*)d_in[1];
  const float* wt = (const float*)d_in[2];
  const float* sw = (const float*)d_in[3];
  const float* sb = (const float*)d_in[4];
  const float* bs = (const float*)d_in[5];
  float* out = (float*)d_out;

  float*    style = (float*)d_ws;                        // 8*512 f32 = 16 KB
  _Float16* wmod  = (_Float16*)((char*)d_ws + 16384);    // 8*512*512 f16 = 4 MB

  style_kernel<<<dim3(8), dim3(512), 0, stream>>>(w, sw, sb, style);
  modw_kernel<<<dim3(512, 8), dim3(256), 0, stream>>>(wt, style, wmod);
  gemm_kernel<<<dim3(4, 128, 8), dim3(256), 0, stream>>>(x, wmod, bs, out);
}
